// NonsharedPatchEmbed_86827058856432
// MI455X (gfx1250) — matrix operand order repaired and checked by reading, not compiled, not run on hardware
//
#include <hip/hip_runtime.h>

typedef __attribute__((ext_vector_type(16))) __bf16 v16bf;
typedef __attribute__((ext_vector_type(8)))  __bf16 v8bf;
typedef __attribute__((ext_vector_type(4)))  __bf16 v4bf;
typedef __attribute__((ext_vector_type(4)))  float  v4f;
typedef __attribute__((ext_vector_type(8)))  float  v8f;

#define BM   64
#define BN   256
#define BK   32
#define LROW 40   // padded LDS row (bf16 elems): 80B pitch, keeps 16B alignment
#define NKT  (768 / BK)

// Convert a register-staged tile to bf16 and store it into one LDS buffer.
__device__ __forceinline__ void stage_tile(const v4f (&wreg)[8], const float (&areg)[8],
                                           __bf16 (*Asb)[LROW], __bf16 (*Wsb)[LROW],
                                           int wrow0, int wkq, int am, int akc) {
#pragma unroll
  for (int pass = 0; pass < 8; ++pass) {
    // packed f32->bf16 (RTNE): lowers to v_cvt_pk_bf16_f32; 8B ds_store
    *(v4bf*)&Wsb[wrow0 + pass * 32][wkq] = __builtin_convertvector(wreg[pass], v4bf);
  }
  v4f a0 = (v4f){areg[0], areg[1], areg[2], areg[3]};
  v4f a1 = (v4f){areg[4], areg[5], areg[6], areg[7]};
  *(v4bf*)&Asb[am][akc]     = __builtin_convertvector(a0, v4bf);
  *(v4bf*)&Asb[am][akc + 4] = __builtin_convertvector(a1, v4bf);
}

__global__ __launch_bounds__(256) void patch_embed_wmma(
    const float* __restrict__ X,   // [64, 3, 224, 224]
    const float* __restrict__ W,   // [196, 768(out), 768(in)]
    const float* __restrict__ Bv,  // [196, 768]
    float* __restrict__ O) {       // [64, 196, 768]
  __shared__ __bf16 As[2][BM][LROW];   //  10.24 KB
  __shared__ __bf16 Ws[2][BN][LROW];   // 40.96 KB  (total 51.2 KB of 320 KB/WGP)

  const int p    = blockIdx.y;
  const int n0   = blockIdx.x * BN;
  const int ph   = p / 14, pw = p % 14;
  const int tid  = threadIdx.x;
  const int lane = tid & 31;
  const int wave = tid >> 5;
  const int h    = lane >> 4;
  const int l16  = lane & 15;
  const int mtile = (wave >> 1) * 16;  // 4 M-tiles, 2 waves each
  const int nbase = (wave & 1) * 128;  // each wave: 8 N-tiles of 16
  const int mrow  = mtile + l16;

  const float* Wp = W + (size_t)p * 768 * 768;

  // Accumulators pre-loaded with bias (bias depends only on column = l16)
  v8f acc[8];
#pragma unroll
  for (int j = 0; j < 8; ++j) {
    float bb = Bv[p * 768 + n0 + nbase + j * 16 + l16];
    acc[j] = (v8f){bb, bb, bb, bb, bb, bb, bb, bb};
  }

  // staging work split
  const int wrow0 = tid >> 3;        // 0..31 (8 passes -> 256 rows)
  const int wkq   = (tid & 7) * 4;   // float4 granularity along K
  const int am    = tid >> 2;        // 0..63
  const int akc   = (tid & 3) * 8;   // 8 k's per thread

  const float* wbase = Wp + (size_t)(n0 + wrow0) * 768 + wkq;

  v4f   wreg[8];
  float areg[8];

  // ---------------- prologue: fetch + stage K-step 0 ----------------
#pragma unroll
  for (int pass = 0; pass < 8; ++pass)
    wreg[pass] = *(const v4f*)(wbase + (size_t)pass * 32 * 768);
#pragma unroll
  for (int i = 0; i < 8; ++i) {
    const int ka = akc + i;                 // k = (pp*16+q)*3 + c
    const int c = ka % 3, t2 = ka / 3, q = t2 & 15, pp = t2 >> 4;
    areg[i] = X[((size_t)(am * 3 + c) * 224 + (ph * 16 + pp)) * 224 + (pw * 16 + q)];
  }
  stage_tile(wreg, areg, As[0], Ws[0], wrow0, wkq, am, akc);
  __syncthreads();

  // ---------------- pipelined main loop ----------------
  for (int kt = 0; kt < NKT; ++kt) {
    const int cur = kt & 1;

    // 1) issue NEXT tile's global loads (latency hides under the WMMA block)
    if (kt + 1 < NKT) {
      const int K1 = (kt + 1) * BK;
#pragma unroll
      for (int pass = 0; pass < 8; ++pass)
        wreg[pass] = *(const v4f*)(wbase + (size_t)pass * 32 * 768 + K1);
#pragma unroll
      for (int i = 0; i < 8; ++i) {
        const int ka = K1 + akc + i;
        const int c = ka % 3, t2 = ka / 3, q = t2 & 15, pp = t2 >> 4;
        areg[i] = X[((size_t)(am * 3 + c) * 224 + (ph * 16 + pp)) * 224 + (pw * 16 + q)];
      }
    }

    // 2) compute on current buffer
    {
      const __bf16 (*Asb)[LROW] = As[cur];
      const __bf16 (*Wsb)[LROW] = Ws[cur];

      // A fragment: lane = M row; K = h*16 + i (matches B)
      v8bf alo = *(const v8bf*)&Asb[mrow][h * 16];
      v8bf ahi = *(const v8bf*)&Asb[mrow][h * 16 + 8];
      const v16bf af = __builtin_shufflevector(alo, ahi, 0, 1, 2, 3, 4, 5, 6, 7,
                                               8, 9, 10, 11, 12, 13, 14, 15);

      // Preload ALL 8 B fragments (lane = N col; K = h*16 + i) into distinct
      // registers: the 16 ds_load_b128s issue as one burst and the per-WMMA
      // waits become staggered partial s_wait_dscnt instead of 8x dscnt==0.
      const __bf16* wsrow = &Wsb[nbase + l16][h * 16];
      v16bf bf[8];
#pragma unroll
      for (int j = 0; j < 8; ++j) {
        const __bf16* r = wsrow + (size_t)j * 16 * LROW;
        v8bf lo = *(const v8bf*)(r);
        v8bf hi = *(const v8bf*)(r + 8);
        bf[j] = __builtin_shufflevector(lo, hi, 0, 1, 2, 3, 4, 5, 6, 7,
                                        8, 9, 10, 11, 12, 13, 14, 15);
      }
#pragma unroll
      for (int j = 0; j < 8; ++j) {
        acc[j] = __builtin_amdgcn_wmma_f32_16x16x32_bf16(
            false, af, false, bf[j], (short)0, acc[j], false, false);
      }
    }

    // 3) convert + store next tile into the alternate buffer
    if (kt + 1 < NKT)
      stage_tile(wreg, areg, As[cur ^ 1], Ws[cur ^ 1], wrow0, wkq, am, akc);

    // 4) one barrier per iteration
    __syncthreads();
  }

  // ---------------- store: C/D layout M = r + 8*h, N = l16 ----------------
#pragma unroll
  for (int j = 0; j < 8; ++j) {
    const int ng = n0 + nbase + j * 16 + l16;
#pragma unroll
    for (int r = 0; r < 8; ++r) {
      const int m = mtile + r + 8 * h;
      O[((size_t)m * 196 + p) * 768 + ng] = acc[j][r];
    }
  }
}

extern "C" void kernel_launch(void* const* d_in, const int* in_sizes, int n_in,
                              void* d_out, int out_size, void* d_ws, size_t ws_size,
                              hipStream_t stream) {
  (void)in_sizes; (void)n_in; (void)out_size; (void)d_ws; (void)ws_size;
  const float* X  = (const float*)d_in[0];  // [64,3,224,224]
  const float* W  = (const float*)d_in[1];  // [196,768,768]
  const float* Bv = (const float*)d_in[2];  // [196,768]
  float*       O  = (float*)d_out;          // [64,196,768]
  dim3 grid(768 / BN, 196);                 // 3 x 196 workgroups
  dim3 block(256);                          // 8 waves (wave32)
  patch_embed_wmma<<<grid, block, 0, stream>>>(X, W, Bv, O);
}
